// Attention_4612794876616
// MI455X (gfx1250) — matrix operand order repaired and checked by reading, not compiled, not run on hardware
//
#include <hip/hip_runtime.h>
#include <hip/hip_bf16.h>
#include <math.h>

typedef __attribute__((ext_vector_type(16))) _Float16 v16h;
typedef __attribute__((ext_vector_type(8)))  _Float16 v8h;
typedef __attribute__((ext_vector_type(8)))  float    v8f;

#define DIM 768
#define NHEAD 12
#define HDIM 64
#define BATCH 2
#define SEQ 2048
#define M_TOT (BATCH * SEQ)     /* 4096 */
#define N_QKV (3 * DIM)         /* 2304 */
#define ATT_SCALE 0.125f        /* 64^-0.5 */

// ---------------------------------------------------------------------------
// WMMA helpers (CDNA5 wave32, 16x16x32 f16 -> f32)
// ---------------------------------------------------------------------------
__device__ __forceinline__ v8f wmma_f16(v16h a, v16h b, v8f c) {
  return __builtin_amdgcn_wmma_f32_16x16x32_f16(
      /*neg_a=*/false, a, /*neg_b=*/false, b,
      /*c_mod=*/(short)0, c, /*reuse_a=*/false, /*reuse_b=*/false);
}

__device__ __forceinline__ v8f zero_v8f() {
  v8f z = {0.f, 0.f, 0.f, 0.f, 0.f, 0.f, 0.f, 0.f};
  return z;
}

// A-matrix fragment (16x32, row-major source, leading dim ld in elements).
// ISA layout: lanes 0-15 hold row=lane, K={0..7,16..23}; lanes 16-31 hold
// row=lane-16, K={8..15,24..31}.  Two 128-bit loads per lane.
__device__ __forceinline__ v16h load_frag_a(const _Float16* base, int ld) {
  const int lane = threadIdx.x & 31;
  const int row  = lane & 15;
  const int hi   = lane >> 4;
  const _Float16* p = base + row * ld + hi * 8;
  v8h c0 = *(const v8h*)(p);
  v8h c1 = *(const v8h*)(p + 16);
  v16h r;
#pragma unroll
  for (int i = 0; i < 8; ++i) { r[i] = c0[i]; r[8 + i] = c1[i]; }
  return r;
}

// B-matrix fragment (32x16).  Source is B^T stored [N][K] row-major with
// leading dim ld.  ISA layout: lane holds col N=lane%16; lanes 0-15 hold
// K={0..7,16..23}, lanes 16-31 hold K={8..15,24..31}.  Two 128-bit loads.
__device__ __forceinline__ v16h load_frag_b(const _Float16* bt, int ld) {
  const int lane = threadIdx.x & 31;
  const int n    = lane & 15;
  const int hi   = lane >> 4;
  const _Float16* p = bt + n * ld + hi * 8;
  v8h c0 = *(const v8h*)(p);
  v8h c1 = *(const v8h*)(p + 16);
  v16h r;
#pragma unroll
  for (int i = 0; i < 8; ++i) { r[i] = c0[i]; r[8 + i] = c1[i]; }
  return r;
}

// Cross-lane reductions within each 16-lane half (C-fragment N dimension).
__device__ __forceinline__ float red_max16(float v) {
#pragma unroll
  for (int m = 1; m < 16; m <<= 1) v = fmaxf(v, __shfl_xor(v, m, 32));
  return v;
}
__device__ __forceinline__ float red_sum16(float v) {
#pragma unroll
  for (int m = 1; m < 16; m <<= 1) v += __shfl_xor(v, m, 32);
  return v;
}

// ---------------------------------------------------------------------------
// Conversion kernels
// ---------------------------------------------------------------------------
__global__ void cvt_f32_f16(const float* __restrict__ in,
                            _Float16* __restrict__ out, int n) {
  int i = blockIdx.x * blockDim.x + threadIdx.x;
  if (i < n) out[i] = (_Float16)in[i];
}

// wt[n*K + k] = (f16) w[k*N + n]   (store W^T, [N][K] row-major)
__global__ void transpose_cvt(const float* __restrict__ w,
                              _Float16* __restrict__ wt, int K, int N) {
  int idx = blockIdx.x * blockDim.x + threadIdx.x;
  if (idx < K * N) {
    int n = idx / K;
    int k = idx - n * K;
    wt[idx] = (_Float16)w[k * N + n];
  }
}

// ---------------------------------------------------------------------------
// QKV GEMM: qkv[M,2304] = xh[M,768] @ Wqkv + bqkv, scattered into
//   Qh [B,H,L,64] (pre-scaled by ATT_SCALE), Kh [B,H,L,64], Vt [B,H,64,L]
// One wave computes a 32(M) x 64(N) tile.  8 waves / block.
// ---------------------------------------------------------------------------
__global__ void qkv_gemm(const _Float16* __restrict__ xh,
                         const _Float16* __restrict__ wt,
                         const float* __restrict__ bias,
                         _Float16* __restrict__ Qh,
                         _Float16* __restrict__ Kh,
                         _Float16* __restrict__ Vt) {
  const int wave = (blockIdx.x * blockDim.x + threadIdx.x) >> 5;
  const int TN = N_QKV / 64;            // 36
  const int tM = wave / TN;
  const int tN = wave - tM * TN;
  const int m0 = tM * 32;
  const int n0 = tN * 64;
  const int lane = threadIdx.x & 31;
  const int nl = lane & 15;
  const int hi = lane >> 4;

  v8f acc[2][4];
#pragma unroll
  for (int i = 0; i < 2; ++i)
#pragma unroll
    for (int j = 0; j < 4; ++j) acc[i][j] = zero_v8f();

  for (int k0 = 0; k0 < DIM; k0 += 32) {
    v16h a0 = load_frag_a(xh + (size_t)m0 * DIM + k0, DIM);
    v16h a1 = load_frag_a(xh + (size_t)(m0 + 16) * DIM + k0, DIM);
#pragma unroll
    for (int t = 0; t < 4; ++t) {
      v16h b = load_frag_b(wt + (size_t)(n0 + t * 16) * DIM + k0, DIM);
      acc[0][t] = wmma_f16(a0, b, acc[0][t]);
      acc[1][t] = wmma_f16(a1, b, acc[1][t]);
    }
  }

#pragma unroll
  for (int mi2 = 0; mi2 < 2; ++mi2) {
#pragma unroll
    for (int t = 0; t < 4; ++t) {
      const int gn = n0 + t * 16 + nl;
      const float bv = bias[gn];
      const int sec = gn / DIM;          // 0=Q 1=K 2=V (uniform per subtile)
      const int c = gn - sec * DIM;
      const int h = c >> 6;              // /64
      const int d = c & 63;
#pragma unroll
      for (int r = 0; r < 8; ++r) {
        const int gm = m0 + mi2 * 16 + r + hi * 8;
        const int bz = gm >> 11;         // / SEQ
        const int i  = gm & (SEQ - 1);
        const float val = acc[mi2][t][r] + bv;
        const size_t hb = (size_t)(bz * NHEAD + h);
        if (sec == 0)
          Qh[(hb * SEQ + i) * HDIM + d] = (_Float16)(val * ATT_SCALE);
        else if (sec == 1)
          Kh[(hb * SEQ + i) * HDIM + d] = (_Float16)val;
        else
          Vt[(hb * HDIM + d) * SEQ + i] = (_Float16)val;
      }
    }
  }
}

// ---------------------------------------------------------------------------
// Flash attention: one wave per 16-query tile, 4 waves / block.
// Per 32-key step: 4 wmmas (S = Q K^T), online softmax, P via LDS re-layout,
// 4 wmmas (O += P V).
// ---------------------------------------------------------------------------
__global__ void attn_kernel(const _Float16* __restrict__ Qh,
                            const _Float16* __restrict__ Kh,
                            const _Float16* __restrict__ Vt,
                            _Float16* __restrict__ Oh) {
  __shared__ _Float16 lds[4 * 16 * 32];

  const int lane = threadIdx.x & 31;
  const int warp = threadIdx.x >> 5;
  const int nl = lane & 15;
  const int hi = lane >> 4;
  const int bh = blockIdx.x;                 // b*NHEAD + h
  const int bz = bh / NHEAD;
  const int h  = bh - bz * NHEAD;
  const int i0 = (blockIdx.y * 4 + warp) * 16;

  const _Float16* Qb = Qh + (size_t)bh * SEQ * HDIM;
  const _Float16* Kb = Kh + (size_t)bh * SEQ * HDIM;
  const _Float16* Vb = Vt + (size_t)bh * HDIM * SEQ;
  _Float16* wlds = &lds[warp * 16 * 32];

  const v16h aq0 = load_frag_a(Qb + (size_t)i0 * HDIM, HDIM);
  const v16h aq1 = load_frag_a(Qb + (size_t)i0 * HDIM + 32, HDIM);

  v8f o[4];
#pragma unroll
  for (int t = 0; t < 4; ++t) o[t] = zero_v8f();
  float mi[8], li[8];
#pragma unroll
  for (int r = 0; r < 8; ++r) { mi[r] = -1e30f; li[r] = 0.f; }

  for (int j0 = 0; j0 < SEQ; j0 += 32) {
    v8f s0 = zero_v8f();
    v8f s1 = zero_v8f();
    s0 = wmma_f16(aq0, load_frag_b(Kb + (size_t)j0 * HDIM, HDIM), s0);
    s0 = wmma_f16(aq1, load_frag_b(Kb + (size_t)j0 * HDIM + 32, HDIM), s0);
    s1 = wmma_f16(aq0, load_frag_b(Kb + (size_t)(j0 + 16) * HDIM, HDIM), s1);
    s1 = wmma_f16(aq1, load_frag_b(Kb + (size_t)(j0 + 16) * HDIM + 32, HDIM), s1);

#pragma unroll
    for (int r = 0; r < 8; ++r) {
      const float rm = red_max16(fmaxf(s0[r], s1[r]));
      const float mnew = fmaxf(mi[r], rm);
      const float corr = __expf(mi[r] - mnew);
      const float p0 = __expf(s0[r] - mnew);
      const float p1 = __expf(s1[r] - mnew);
      const float rs = red_sum16(p0 + p1);
      li[r] = li[r] * corr + rs;
      mi[r] = mnew;
#pragma unroll
      for (int t = 0; t < 4; ++t) o[t][r] *= corr;
      const int row = r + hi * 8;
      wlds[row * 32 + nl]      = (_Float16)p0;
      wlds[row * 32 + nl + 16] = (_Float16)p1;
    }
    __syncthreads();
    const v16h pf = load_frag_a(wlds, 32);
#pragma unroll
    for (int t = 0; t < 4; ++t)
      o[t] = wmma_f16(pf, load_frag_b(Vb + (size_t)(t * 16) * SEQ + j0, SEQ),
                      o[t]);
    __syncthreads();
  }

#pragma unroll
  for (int r = 0; r < 8; ++r) {
    const float inv = 1.0f / li[r];
    const int gi = i0 + r + hi * 8;
    const size_t rowbase = ((size_t)(bz * SEQ + gi)) * DIM + h * HDIM;
#pragma unroll
    for (int t = 0; t < 4; ++t)
      Oh[rowbase + t * 16 + nl] = (_Float16)(o[t][r] * inv);
  }
}

// ---------------------------------------------------------------------------
// Output projection: out[M,768] = Oh @ Wproj + bproj   (fp32 output)
// ---------------------------------------------------------------------------
__global__ void proj_gemm(const _Float16* __restrict__ Oh,
                          const _Float16* __restrict__ wt,
                          const float* __restrict__ bias,
                          float* __restrict__ out) {
  const int wave = (blockIdx.x * blockDim.x + threadIdx.x) >> 5;
  const int TN = DIM / 64;              // 12
  const int tM = wave / TN;
  const int tN = wave - tM * TN;
  const int m0 = tM * 32;
  const int n0 = tN * 64;
  const int lane = threadIdx.x & 31;
  const int nl = lane & 15;
  const int hi = lane >> 4;

  v8f acc[2][4];
#pragma unroll
  for (int i = 0; i < 2; ++i)
#pragma unroll
    for (int j = 0; j < 4; ++j) acc[i][j] = zero_v8f();

  for (int k0 = 0; k0 < DIM; k0 += 32) {
    v16h a0 = load_frag_a(Oh + (size_t)m0 * DIM + k0, DIM);
    v16h a1 = load_frag_a(Oh + (size_t)(m0 + 16) * DIM + k0, DIM);
#pragma unroll
    for (int t = 0; t < 4; ++t) {
      v16h b = load_frag_b(wt + (size_t)(n0 + t * 16) * DIM + k0, DIM);
      acc[0][t] = wmma_f16(a0, b, acc[0][t]);
      acc[1][t] = wmma_f16(a1, b, acc[1][t]);
    }
  }

#pragma unroll
  for (int mi2 = 0; mi2 < 2; ++mi2) {
#pragma unroll
    for (int t = 0; t < 4; ++t) {
      const int gn = n0 + t * 16 + nl;
      const float bv = bias[gn];
#pragma unroll
      for (int r = 0; r < 8; ++r) {
        const int gm = m0 + mi2 * 16 + r + hi * 8;
        out[(size_t)gm * DIM + gn] = acc[mi2][t][r] + bv;
      }
    }
  }
}

// ---------------------------------------------------------------------------
// Launch
// ---------------------------------------------------------------------------
extern "C" void kernel_launch(void* const* d_in, const int* in_sizes, int n_in,
                              void* d_out, int out_size, void* d_ws,
                              size_t ws_size, hipStream_t stream) {
  (void)in_sizes; (void)n_in; (void)out_size; (void)ws_size;

  const float* x     = (const float*)d_in[0];
  const float* Wqkv  = (const float*)d_in[1];
  const float* bqkv  = (const float*)d_in[2];
  const float* Wproj = (const float*)d_in[3];
  const float* bproj = (const float*)d_in[4];
  float* out = (float*)d_out;

  char* ws = (char*)d_ws;
  // f16 workspace layout (bytes)
  _Float16* xh     = (_Float16*)(ws);                 // 4096*768*2  = 6291456
  _Float16* wqkvT  = (_Float16*)(ws + 6291456);       // 2304*768*2  = 3538944
  _Float16* wprojT = (_Float16*)(ws + 9830400);       //  768*768*2  = 1179648
  _Float16* Qh     = (_Float16*)(ws + 11010048);      // 6291456
  _Float16* Kh     = (_Float16*)(ws + 17301504);      // 6291456
  _Float16* Vt     = (_Float16*)(ws + 23592960);      // 6291456
  _Float16* Oh     = (_Float16*)(ws + 29884416);      // 6291456  (tot ~36.2MB)

  cvt_f32_f16<<<(M_TOT * DIM + 255) / 256, 256, 0, stream>>>(x, xh,
                                                             M_TOT * DIM);
  transpose_cvt<<<(DIM * N_QKV + 255) / 256, 256, 0, stream>>>(Wqkv, wqkvT,
                                                               DIM, N_QKV);
  transpose_cvt<<<(DIM * DIM + 255) / 256, 256, 0, stream>>>(Wproj, wprojT,
                                                             DIM, DIM);

  // 4608 wave-tiles of 32x64; 8 waves per 256-thread block -> 576 blocks
  qkv_gemm<<<(M_TOT / 32) * (N_QKV / 64) / 8, 256, 0, stream>>>(
      xh, wqkvT, bqkv, Qh, Kh, Vt);

  // (B*H, SEQ/64) blocks, 4 waves (16-query tiles each) per block
  attn_kernel<<<dim3(BATCH * NHEAD, SEQ / 64), 128, 0, stream>>>(Qh, Kh, Vt,
                                                                 Oh);

  // 1536 wave-tiles -> 192 blocks
  proj_gemm<<<(M_TOT / 32) * (DIM / 64) / 8, 256, 0, stream>>>(Oh, wprojT,
                                                               bproj, out);
}
